// MoELayer_14448269984009
// MI455X (gfx1250) — matrix operand order repaired and checked by reading, not compiled, not run on hardware
//
#include <hip/hip_runtime.h>
#include <stdint.h>
#include <math.h>

// ---------------- problem constants ----------------
#define B_ 4
#define T_ 2048
#define D_ 1024
#define H_ 2048
#define O_ 1024
#define E_ 8
#define NTOK (B_ * T_)          // 8192 tokens
#define LCAP (NTOK + 32)        // per-expert token-list capacity (padded)

// ---------------- types ----------------
typedef __bf16 bf16_t;
typedef __attribute__((ext_vector_type(16))) __bf16 v16bf;
typedef __attribute__((ext_vector_type(8)))  __bf16 v8bf;
typedef __attribute__((ext_vector_type(8)))  float  v8f;

// ---------------- moe gemm tiling ----------------
#define MT        32            // tokens per block tile (2 WMMA M-subtiles)
#define THREADS   512           // 16 wave32s
#define TILES_MAX 256           // ceil(8192 / MT)
#define XB_ELEMS  (MT * D_)     // 32 x 1024 bf16 (64 KB)
#define HB_ELEMS  (MT * H_)     // 32 x 2048 bf16 (128 KB)
#define CHUNK_ELEMS (32 * 512)  // one staged weight chunk: K=32 x 512 cols (32 KB)
#define CHUNK_BYTES (CHUNK_ELEMS * 2)
#define NCHUNK1   128           // GEMM1: 32 k-chunks x 4 col-groups
#define NCHUNK2   128           // GEMM2: 64 k-chunks x 2 col-groups

// ---------------- workspace layout (bytes) ----------------
#define SEL_OFF   0u
#define GV_OFF    (SEL_OFF + (unsigned)(NTOK * 2 * sizeof(int)))      // 65536
#define CNT_OFF   (GV_OFF  + (unsigned)(NTOK * 2 * sizeof(float)))   // 131072
#define LIST_OFF  (CNT_OFF + 1024u)                                   // 132096
#define LG_OFF    (LIST_OFF + (unsigned)(E_ * LCAP * sizeof(int)))    // 395264
#define XBF_OFF   (LG_OFF + (unsigned)(E_ * LCAP * sizeof(float)))    // 658432 (16B aligned)
#define W1B_OFF   (XBF_OFF + (unsigned)(NTOK * D_ * 2))               // +16 MB
#define W2B_OFF   (W1B_OFF + (unsigned)(E_ * D_ * H_ * 2))            // +32 MB

// =====================================================================
// CDNA5 async copy helpers (ASYNCcnt-tracked DMA into LDS)
// =====================================================================
__device__ __forceinline__ void async_copy_b128(uint32_t lds_addr,
                                                unsigned long long gaddr) {
  asm volatile("global_load_async_to_lds_b128 %0, %1, off"
               :: "v"(lds_addr), "v"(gaddr) : "memory");
}
__device__ __forceinline__ void wait_async0() {
  asm volatile("s_wait_asynccnt 0x0" ::: "memory");
}
// stage one contiguous 32 KB chunk: 4 x b128 per thread (512 threads)
__device__ __forceinline__ void stage_chunk(uint32_t lds_base, const char* gsrc,
                                            int tid) {
#pragma unroll
  for (int i = 0; i < 4; ++i) {
    uint32_t u = (uint32_t)(tid + i * THREADS);
    async_copy_b128(lds_base + u * 16u,
                    (unsigned long long)(uintptr_t)gsrc + (unsigned long long)u * 16ull);
  }
}

// =====================================================================
// 0) zero the output (atomic accumulation target)
// =====================================================================
__global__ void zero_f32(float* __restrict__ p, int n) {
  int i = blockIdx.x * blockDim.x + threadIdx.x;
  int stride = gridDim.x * blockDim.x;
  for (; i < n; i += stride) p[i] = 0.0f;
}

// =====================================================================
// 0b) one-time f32 -> bf16 conversions into workspace
// =====================================================================
__global__ void convert_x_kernel(const float* __restrict__ src,
                                 bf16_t* __restrict__ dst, int n) {
  int i = blockIdx.x * blockDim.x + threadIdx.x;
  int stride = gridDim.x * blockDim.x;
  for (; i < n; i += stride) dst[i] = (bf16_t)src[i];
}

// Pre-swizzle weights into the exact WMMA B-fragment LDS image so the hot
// kernel can DMA chunks verbatim. Source: [E][K][N] f32, K*N = 2^21/expert.
// Chunk = K-slab of 32 rows x 512-col group, 16384 bf16. Within chunk:
// tile tn = cc>>4 (16 cols), owner lane = n + ((r>>4)<<4), slot = r&15.
__global__ void convert_w_kernel(const float* __restrict__ src,
                                 bf16_t* __restrict__ dst, int logN) {
  const int total = E_ << 21;
  const int nGrp = 1 << (logN - 9);           // 512-col groups per row
  const int kcPerE = 1 << (21 - logN - 5);    // K/32
  int i = blockIdx.x * blockDim.x + threadIdx.x;
  int stride = gridDim.x * blockDim.x;
  for (; i < total; i += stride) {
    int e = i >> 21;
    int rem = i & ((1 << 21) - 1);
    int k = rem >> logN;
    int c = rem & ((1 << logN) - 1);
    int kc = k >> 5, r = k & 31;
    int grp = c >> 9, cc = c & 511;
    int tn = cc >> 4, nn = cc & 15;
    int lanei = nn + ((r >> 4) << 4);
    int slot = r & 15;
    size_t off = ((size_t)((e * kcPerE + kc) * nGrp + grp) << 14)
               + (size_t)(tn << 9) + (size_t)(lanei << 4) + (size_t)slot;
    dst[off] = (bf16_t)src[i];
  }
}

// =====================================================================
// 1) gating: one wave per token; top-2 of softmax == top-2 logits,
//    renormalized gates g0 = sigmoid(z0 - z1), g1 = 1 - g0.
// =====================================================================
__global__ __launch_bounds__(256) void gate_kernel(
    const float* __restrict__ x, const float* __restrict__ gw,
    const float* __restrict__ gb, int* __restrict__ sel, float* __restrict__ gv)
{
  const int lane = threadIdx.x & 31;
  const int wave = threadIdx.x >> 5;
  const int t = blockIdx.x * 8 + wave;
  if (t >= NTOK) return;

  float acc[E_];
#pragma unroll
  for (int e = 0; e < E_; ++e) acc[e] = 0.0f;

  const float* xr = x + (size_t)t * D_;
  for (int d = lane; d < D_; d += 32) {
    float xv = xr[d];
#pragma unroll
    for (int e = 0; e < E_; ++e) acc[e] += xv * gw[d * E_ + e];
  }
#pragma unroll
  for (int e = 0; e < E_; ++e) {
#pragma unroll
    for (int off = 16; off >= 1; off >>= 1)
      acc[e] += __shfl_xor(acc[e], off, 32);
  }
  if (lane == 0) {
    float z[E_];
#pragma unroll
    for (int e = 0; e < E_; ++e) z[e] = acc[e] + gb[e];
    int i0 = 0;
#pragma unroll
    for (int e = 1; e < E_; ++e) if (z[e] > z[i0]) i0 = e;      // ties: lowest idx
    int i1 = (i0 == 0) ? 1 : 0;
#pragma unroll
    for (int e = 0; e < E_; ++e) if (e != i0 && z[e] > z[i1]) i1 = e;
    float g0 = 1.0f / (1.0f + expf(z[i1] - z[i0]));
    sel[t * 2 + 0] = i0;  sel[t * 2 + 1] = i1;
    gv[t * 2 + 0] = g0;   gv[t * 2 + 1] = 1.0f - g0;
  }
}

// =====================================================================
// 2) routing: one block per expert, deterministic ordered compaction.
// =====================================================================
__global__ __launch_bounds__(256) void route_kernel(
    const int* __restrict__ sel, const float* __restrict__ gv,
    int* __restrict__ cnt, int* __restrict__ list, float* __restrict__ lg)
{
  const int e = blockIdx.x;
  __shared__ int wave_sums[8];
  __shared__ int running;
  const int tid = threadIdx.x, lane = tid & 31, wave = tid >> 5;
  if (tid == 0) running = 0;
  __syncthreads();

  for (int base = 0; base < NTOK; base += 256) {
    int t = base + tid;
    int flag = 0; float g = 0.0f;
    if (sel[t * 2 + 0] == e)      { flag = 1; g = gv[t * 2 + 0]; }
    else if (sel[t * 2 + 1] == e) { flag = 1; g = gv[t * 2 + 1]; }

    unsigned long long m = __ballot(flag);               // wave32: low 32 bits
    int prefix = __popcll(m & ((1ull << lane) - 1ull));
    if (lane == 0) wave_sums[wave] = __popcll(m);
    __syncthreads();

    int wbase = 0;
    for (int w = 0; w < wave; ++w) wbase += wave_sums[w];
    int cbase = running;
    if (flag) {
      int pos = cbase + wbase + prefix;
      list[e * LCAP + pos] = t;
      lg[e * LCAP + pos]   = g;
    }
    __syncthreads();
    if (tid == 0) {
      int tot = 0;
      for (int w = 0; w < 8; ++w) tot += wave_sums[w];
      running += tot;
    }
    __syncthreads();
  }
  int c = running;
  if (tid == 0) cnt[e] = c;
  int padded = (c + (MT - 1)) & ~(MT - 1);
  for (int p = c + tid; p < padded; p += 256) {          // pad: token 0, gate 0
    list[e * LCAP + p] = 0;
    lg[e * LCAP + p]   = 0.0f;
  }
}

// =====================================================================
// 3) fused per-expert MLP: async-DMA staged weights + WMMA bf16
// =====================================================================
// A-fragment (16x32 bf16) from row-major bf16 LDS tile, contiguous k order (same as packed B):
// lane L: M = L&15, K = (L>>4)*16 + i
__device__ __forceinline__ v16bf load_a_frag(const bf16_t* base, int row_stride,
                                             int row0, int k0, int lane)
{
  int r = row0 + (lane & 15);
  const bf16_t* p = base + r * row_stride + k0 + ((lane >> 4) * 16);
  v8bf lo = *(const v8bf*)p;
  v8bf hi = *(const v8bf*)(p + 8);
  v16bf a;
#pragma unroll
  for (int i = 0; i < 8; ++i) { a[i] = lo[i]; a[i + 8] = hi[i]; }
  return a;
}

__global__ __launch_bounds__(THREADS) void moe_expert_kernel(
    const bf16_t* __restrict__ xb16, const bf16_t* __restrict__ w1b,
    const float* __restrict__ b1,    const bf16_t* __restrict__ w2b,
    const float* __restrict__ b2,    const int* __restrict__ cnt,
    const int* __restrict__ list,    const float* __restrict__ lg,
    float* __restrict__ out)
{
  const int e    = blockIdx.x >> 8;        // / TILES_MAX
  const int tile = blockIdx.x & 255;
  if (tile * MT >= cnt[e]) return;          // uniform early exit
  const int row_base = tile * MT;

  extern __shared__ __align__(16) char smem[];
  bf16_t* xb    = (bf16_t*)smem;            // [32][1024] bf16  (64 KB)
  bf16_t* hb    = xb + XB_ELEMS;            // [32][2048] bf16  (128 KB)
  bf16_t* wbuf0 = hb + HB_ELEMS;            // chunk buffer 0   (32 KB)
  bf16_t* wbuf1 = wbuf0 + CHUNK_ELEMS;      // chunk buffer 1   (32 KB)

  const int tid  = threadIdx.x;
  const int lane = tid & 31;
  const int wave = tid >> 5;

  const uint32_t xb_lds  = (uint32_t)(uintptr_t)(void*)xb;
  const uint32_t wb_lds0 = (uint32_t)(uintptr_t)(void*)wbuf0;
  const uint32_t wb_lds1 = (uint32_t)(uintptr_t)(void*)wbuf1;

  const int*   mylist = list + e * LCAP + row_base;
  const float* mylg   = lg   + e * LCAP + row_base;

  // ---- issue X-tile gather: 32 token rows x 2 KB, per-lane async B128 ----
#pragma unroll
  for (int i = 0; i < 8; ++i) {
    int u = tid + i * THREADS;              // 4096 copy units of 16 B
    int r = u >> 7;                         // 128 units per 2 KB row
    unsigned long long ga =
        (unsigned long long)(uintptr_t)(xb16 + (size_t)mylist[r] * D_)
        + (unsigned long long)(u & 127) * 16ull;
    async_copy_b128(xb_lds + (uint32_t)u * 16u, ga);
  }

  const char* w1base = (const char*)(w1b + (size_t)e * D_ * H_); // 128 chunks
  const char* w2base = (const char*)(w2b + (size_t)e * H_ * O_); // 128 chunks

  // ---- prime the weight pipeline ----
  stage_chunk(wb_lds0, w1base, tid);

  v8f zero8 = {0.f, 0.f, 0.f, 0.f, 0.f, 0.f, 0.f, 0.f};

  // ---- GEMM1: h = relu(X @ W1[e] + b1[e]) ----
  // chunk c = kc*4 + grp covers K rows kc*32..+31, cols grp*512..+511.
  // wave owns 32 cols per chunk -> acc1[grp][nt][msub]
  v8f acc1[4][2][2];
#pragma unroll
  for (int g = 0; g < 4; ++g)
#pragma unroll
    for (int nt = 0; nt < 2; ++nt)
#pragma unroll
      for (int ms = 0; ms < 2; ++ms) acc1[g][nt][ms] = zero8;

  for (int kc = 0; kc < 32; ++kc) {
    v16bf af0, af1;
#pragma unroll
    for (int grp = 0; grp < 4; ++grp) {
      const int c   = kc * 4 + grp;
      const int cur = c & 1;
      wait_async0();                        // chunk c (and X tile on c==0) landed
      __syncthreads();                      // publish LDS to all waves
      if (c + 1 < NCHUNK1)
        stage_chunk(cur ? wb_lds0 : wb_lds1, w1base + (size_t)(c + 1) * CHUNK_BYTES, tid);
      else
        stage_chunk(cur ? wb_lds0 : wb_lds1, w2base, tid);   // bridge into GEMM2
      if (grp == 0) {
        af0 = load_a_frag(xb, D_, 0,  kc * 32, lane);
        af1 = load_a_frag(xb, D_, 16, kc * 32, lane);
      }
      const bf16_t* wb = cur ? wbuf1 : wbuf0;
#pragma unroll
      for (int nt = 0; nt < 2; ++nt) {
        v16bf bfr = *(const v16bf*)(wb + (wave * 2 + nt) * 512 + lane * 16);
        acc1[grp][nt][0] = __builtin_amdgcn_wmma_f32_16x16x32_bf16(
            false, af0, false, bfr, (short)0, acc1[grp][nt][0], false, false);
        acc1[grp][nt][1] = __builtin_amdgcn_wmma_f32_16x16x32_bf16(
            false, af1, false, bfr, (short)0, acc1[grp][nt][1], false, false);
      }
    }
  }

  // ---- epilogue1: bias + relu, f32 -> bf16, write h tile ----
  // D layout: lane L -> col = L&15, rows = (L>>4)*8 + v
  const float* b1e = b1 + e * H_;
#pragma unroll
  for (int grp = 0; grp < 4; ++grp)
#pragma unroll
    for (int nt = 0; nt < 2; ++nt) {
      int n = grp * 512 + wave * 32 + nt * 16 + (lane & 15);
      float bias = b1e[n];
#pragma unroll
      for (int ms = 0; ms < 2; ++ms) {
        int mb = ms * 16 + ((lane >> 4) << 3);
#pragma unroll
        for (int v = 0; v < 8; ++v) {
          float hv = acc1[grp][nt][ms][v] + bias;
          hb[(mb + v) * H_ + n] = (bf16_t)(hv > 0.0f ? hv : 0.0f);
        }
      }
    }

  // ---- GEMM2: y = h @ W2[e]; chunk c = kc*2 + grp ----
  v8f acc2[2][2][2];
#pragma unroll
  for (int g = 0; g < 2; ++g)
#pragma unroll
    for (int nt = 0; nt < 2; ++nt)
#pragma unroll
      for (int ms = 0; ms < 2; ++ms) acc2[g][nt][ms] = zero8;

  for (int kc = 0; kc < 64; ++kc) {
    v16bf af0, af1;
#pragma unroll
    for (int grp = 0; grp < 2; ++grp) {
      const int c   = kc * 2 + grp;
      const int cur = c & 1;                // parity continues from GEMM1 bridge
      wait_async0();
      __syncthreads();                      // also publishes hb on c==0
      if (c + 1 < NCHUNK2)
        stage_chunk(cur ? wb_lds0 : wb_lds1, w2base + (size_t)(c + 1) * CHUNK_BYTES, tid);
      if (grp == 0) {
        af0 = load_a_frag(hb, H_, 0,  kc * 32, lane);
        af1 = load_a_frag(hb, H_, 16, kc * 32, lane);
      }
      const bf16_t* wb = cur ? wbuf1 : wbuf0;
#pragma unroll
      for (int nt = 0; nt < 2; ++nt) {
        v16bf bfr = *(const v16bf*)(wb + (wave * 2 + nt) * 512 + lane * 16);
        acc2[grp][nt][0] = __builtin_amdgcn_wmma_f32_16x16x32_bf16(
            false, af0, false, bfr, (short)0, acc2[grp][nt][0], false, false);
        acc2[grp][nt][1] = __builtin_amdgcn_wmma_f32_16x16x32_bf16(
            false, af1, false, bfr, (short)0, acc2[grp][nt][1], false, false);
      }
    }
  }

  // ---- epilogue2: + b2, scale by gate, atomic accumulate ----
  const float* b2e = b2 + e * O_;
#pragma unroll
  for (int grp = 0; grp < 2; ++grp)
#pragma unroll
    for (int nt = 0; nt < 2; ++nt) {
      int n = grp * 512 + wave * 32 + nt * 16 + (lane & 15);
      float bias = b2e[n];
#pragma unroll
      for (int ms = 0; ms < 2; ++ms) {
        int mb = ms * 16 + ((lane >> 4) << 3);
#pragma unroll
        for (int v = 0; v < 8; ++v) {
          int rs = mb + v;
          int tok = mylist[rs];
          float g = mylg[rs];               // padded rows: g == 0
          atomicAdd(out + (size_t)tok * O_ + n, g * (acc2[grp][nt][ms][v] + bias));
        }
      }
    }
}

// =====================================================================
// launch
// =====================================================================
extern "C" void kernel_launch(void* const* d_in, const int* in_sizes, int n_in,
                              void* d_out, int out_size, void* d_ws, size_t ws_size,
                              hipStream_t stream)
{
  (void)in_sizes; (void)n_in; (void)ws_size;
  const float* x  = (const float*)d_in[0];
  const float* gw = (const float*)d_in[1];
  const float* gb = (const float*)d_in[2];
  const float* w1 = (const float*)d_in[3];
  const float* b1 = (const float*)d_in[4];
  const float* w2 = (const float*)d_in[5];
  const float* b2 = (const float*)d_in[6];
  // d_in[7] = num_experts_per_tok (fixed top-2 here)
  float* out = (float*)d_out;

  char* ws = (char*)d_ws;
  int*    sel  = (int*)   (ws + SEL_OFF);
  float*  gv   = (float*) (ws + GV_OFF);
  int*    cnt  = (int*)   (ws + CNT_OFF);
  int*    list = (int*)   (ws + LIST_OFF);
  float*  lgt  = (float*) (ws + LG_OFF);
  bf16_t* xbf  = (bf16_t*)(ws + XBF_OFF);
  bf16_t* w1b  = (bf16_t*)(ws + W1B_OFF);
  bf16_t* w2b  = (bf16_t*)(ws + W2B_OFF);

  zero_f32<<<4096, 256, 0, stream>>>(out, out_size);
  convert_x_kernel<<<4096, 256, 0, stream>>>(x, xbf, NTOK * D_);
  convert_w_kernel<<<8192, 256, 0, stream>>>(w1, w1b, 11);   // [E][1024][2048]
  convert_w_kernel<<<8192, 256, 0, stream>>>(w2, w2b, 10);   // [E][2048][1024]
  gate_kernel<<<NTOK / 8, 256, 0, stream>>>(x, gw, gb, sel, gv);
  route_kernel<<<E_, 256, 0, stream>>>(sel, gv, cnt, list, lgt);

  const size_t lds_bytes =
      (size_t)(XB_ELEMS + HB_ELEMS + 2 * CHUNK_ELEMS) * sizeof(bf16_t); // 256 KB
  (void)hipFuncSetAttribute((const void*)moe_expert_kernel,
                            hipFuncAttributeMaxDynamicSharedMemorySize,
                            (int)lds_bytes);
  moe_expert_kernel<<<E_ * TILES_MAX, THREADS, lds_bytes, stream>>>(
      xbf, w1b, b1, w2b, b2, cnt, list, lgt, out);
}
